// MyDecoderLayer_44650480009705
// MI455X (gfx1250) — matrix operand order repaired and checked by reading, not compiled, not run on hardware
//
#include <hip/hip_runtime.h>

// ---------------------------------------------------------------------------
// Types
// ---------------------------------------------------------------------------
typedef __bf16 v16bf __attribute__((ext_vector_type(16)));
typedef float  v8f   __attribute__((ext_vector_type(8)));
typedef unsigned short u16t;
typedef unsigned int   u32t;

union FragAB { v16bf v; u32t u[8]; };

__device__ __forceinline__ u16t f2bf(float f) {
    u32t u = __float_as_uint(f);
    u32t r = u + 0x7fffu + ((u >> 16) & 1u);   // round-to-nearest-even
    return (u16t)(r >> 16);
}

// ---------------------------------------------------------------------------
// CDNA5 async global->LDS copy (GLOBAL_LOAD_ASYNC_TO_LDS_B128, ASYNCcnt).
// Builtin signature (from compiler diagnostic): takes pointers to
// int __attribute__((vector_size(16))) in addrspace(1) / addrspace(3).
// Guarded: falls back to sync load+ds_store staging if builtins are absent.
// ---------------------------------------------------------------------------
#if defined(__has_builtin)
#if __has_builtin(__builtin_amdgcn_global_load_async_to_lds_b128) && \
    __has_builtin(__builtin_amdgcn_s_wait_asynccnt)
#define USE_ASYNC_LDS 1
#endif
#endif
#ifndef USE_ASYNC_LDS
#define USE_ASYNC_LDS 0
#endif

#if USE_ASYNC_LDS
typedef int vsi4 __attribute__((vector_size(16)));
typedef __attribute__((address_space(1))) vsi4 gas_v4i;   // global
typedef __attribute__((address_space(3))) vsi4 las_v4i;   // LDS

__device__ __forceinline__ void async_cp16(const void* g, void* l) {
    __builtin_amdgcn_global_load_async_to_lds_b128(
        (gas_v4i*)(unsigned long long)g,
        (las_v4i*)(unsigned int)(unsigned long long)l,
        0, 0);
}
#endif

// ---------------------------------------------------------------------------
// f32 -> bf16 convert (grid-stride)
// ---------------------------------------------------------------------------
__global__ void k_cvt_f32_bf16(const float* __restrict__ src,
                               u16t* __restrict__ dst, long long n) {
    long long i = (long long)blockIdx.x * blockDim.x + threadIdx.x;
    long long stride = (long long)gridDim.x * blockDim.x;
    for (; i < n; i += stride) dst[i] = f2bf(src[i]);
}

// ---------------------------------------------------------------------------
// Tiled WMMA GEMM:  C[M,N] = alpha * A[M,K](bf16) @ B(bf16) (+bias) (+GELU)
//   TRANSB=0: B element (k,n) = Bmem[k*ldb + n]   (N may be < tile, guarded)
//   TRANSB=1: B element (k,n) = Bmem[n*ldb + k]   (requires N % 128 == 0)
// REQUIRES: M % 128 == 0, K % 32 == 0 (all call sites satisfy this).
// Block tile 128x128, K-step 32, double-buffered LDS async pipeline,
// 8 waves (4x2), wave tile 32x64 -> 8x v_wmma_f32_16x16x32_bf16 per K-step.
// ---------------------------------------------------------------------------
template <bool TRANSB, bool GELU>
__global__ __launch_bounds__(256) void k_gemm_bf16(
    const u16t* __restrict__ A, long long sA, int lda,
    const u16t* __restrict__ B, long long sB, int ldb,
    float* __restrict__ Cf, u16t* __restrict__ Cb, long long sC, int ldc,
    int M, int N, int K, float alpha, const float* __restrict__ bias) {

    __shared__ u16t As[2][128 * 40];   // [row][k], stride 40 keeps 16B align
    __shared__ u16t Bs[2][128 * 40];   // [n][k]   (transposed staging)

    const int tid  = threadIdx.x;
    const int lane = tid & 31, wid = tid >> 5;
    const int wr = wid >> 1, wc = wid & 1;
    const int half = lane >> 4, lm = lane & 15;
    const int bm = blockIdx.y * 128, bn = blockIdx.x * 128;
    const int bz = blockIdx.z;
    const bool fullN = (bn + 128 <= N);

    A += (long long)bz * sA;
    B += (long long)bz * sB;
    float* Cfp = Cf ? Cf + (long long)bz * sC : nullptr;
    u16t*  Cbp = Cb ? Cb + (long long)bz * sC : nullptr;

    // ---- stage one 128x32 K-tile (A and B) into LDS buffer p ----
    auto stage = [&](int kk, int p) {
#pragma unroll
        for (int i = 0; i < 2; ++i) {           // A: 512 chunks of 16B
            int chunk = tid + i * 256;
            int r = chunk >> 2, c8 = (chunk & 3) * 8;
            const u16t* gsrc = A + (long long)(bm + r) * lda + kk + c8;
            u16t* ldst = &As[p][r * 40 + c8];
#if USE_ASYNC_LDS
            async_cp16(gsrc, ldst);
#else
            *(uint4*)ldst = *(const uint4*)gsrc;
#endif
        }
        if (TRANSB) {
#pragma unroll
            for (int i = 0; i < 2; ++i) {       // B rows are n, cols are k
                int chunk = tid + i * 256;
                int r = chunk >> 2, c8 = (chunk & 3) * 8;
                const u16t* gsrc = B + (long long)(bn + r) * ldb + kk + c8;
                u16t* ldst = &Bs[p][r * 40 + c8];
#if USE_ASYNC_LDS
                async_cp16(gsrc, ldst);
#else
                *(uint4*)ldst = *(const uint4*)gsrc;
#endif
            }
        } else {
            // scatter-transpose: mem [k][n] -> LDS [n][k]
#pragma unroll
            for (int i = 0; i < 2; ++i) {
                int chunk = tid + i * 256;
                int kr = chunk >> 4, n8 = (chunk & 15) * 8;
                const u16t* src = B + (long long)(kk + kr) * ldb + bn + n8;
                u16t tmp[8];
                if (fullN || (bn + n8 + 8 <= N)) {
                    *(uint4*)tmp = *(const uint4*)src;
                } else {
#pragma unroll
                    for (int j = 0; j < 8; ++j)
                        tmp[j] = (bn + n8 + j < N) ? src[j] : (u16t)0;
                }
#pragma unroll
                for (int j = 0; j < 8; ++j)
                    Bs[p][(n8 + j) * 40 + kr] = tmp[j];
            }
        }
    };

    v8f acc[2][4];
#pragma unroll
    for (int t = 0; t < 2; ++t)
#pragma unroll
        for (int u = 0; u < 4; ++u)
#pragma unroll
            for (int j = 0; j < 8; ++j) acc[t][u][j] = 0.0f;

    stage(0, 0);                                   // prologue: tile 0

    for (int k0 = 0; k0 < K; k0 += 32) {
        const int p = (k0 >> 5) & 1;
#if USE_ASYNC_LDS
        __builtin_amdgcn_s_wait_asynccnt(0);       // my async writes done
#else
        if (k0 + 32 < K) {                         // L2 prefetch fallback
            int r = tid >> 2, c8 = (tid & 3) * 8;
            __builtin_prefetch(A + (long long)(bm + r) * lda + k0 + 32 + c8, 0, 1);
        }
#endif
        __syncthreads();                           // everyone's writes done
        if (k0 + 32 < K) stage(k0 + 32, p ^ 1);    // overlap next tile copy

        // ---- fragments from buffer p ----
        FragAB af[2], bf[4];
#pragma unroll
        for (int t = 0; t < 2; ++t) {
            int row = wr * 32 + t * 16 + lm;
            const u32t* p0 = (const u32t*)&As[p][row * 40 + half * 8];
            const u32t* p1 = (const u32t*)&As[p][row * 40 + half * 8 + 16];
#pragma unroll
            for (int j = 0; j < 4; ++j) { af[t].u[j] = p0[j]; af[t].u[4 + j] = p1[j]; }
        }
#pragma unroll
        for (int u = 0; u < 4; ++u) {
            int n = wc * 64 + u * 16 + lm;
            const u32t* pp = (const u32t*)&Bs[p][n * 40 + half * 8];
            const u32t* pq = (const u32t*)&Bs[p][n * 40 + half * 8 + 16];
#pragma unroll
            for (int j = 0; j < 4; ++j) { bf[u].u[j] = pp[j]; bf[u].u[4 + j] = pq[j]; }
        }
#pragma unroll
        for (int t = 0; t < 2; ++t)
#pragma unroll
            for (int u = 0; u < 4; ++u)
                acc[t][u] = __builtin_amdgcn_wmma_f32_16x16x32_bf16(
                    false, af[t].v, false, bf[u].v, (short)0, acc[t][u],
                    false, false);
    }

    // ---- epilogue: alpha, bias, GELU, dual f32/bf16 store ----
    auto epi = [&](bool guard) {
#pragma unroll
        for (int t = 0; t < 2; ++t)
#pragma unroll
            for (int u = 0; u < 4; ++u)
#pragma unroll
                for (int i = 0; i < 8; ++i) {
                    int gr = bm + wr * 32 + t * 16 + i + 8 * half;
                    int gc = bn + wc * 64 + u * 16 + lm;
                    if (!guard || gc < N) {
                        float v = acc[t][u][i] * alpha;
                        if (bias) v += bias[gc];
                        if (GELU) v = 0.5f * v * (1.0f + erff(v * 0.70710678118f));
                        if (Cfp) Cfp[(long long)gr * ldc + gc] = v;
                        if (Cbp) Cbp[(long long)gr * ldc + gc] = f2bf(v);
                    }
                }
    };
    if (fullN) epi(false); else epi(true);
}

// ---------------------------------------------------------------------------
// Row softmax: f32 logits -> bf16 probabilities. One block per row.
// ---------------------------------------------------------------------------
__global__ __launch_bounds__(256) void k_softmax_rows(
    const float* __restrict__ X, u16t* __restrict__ Yb, int C) {
    __shared__ float red[256];
    const int tid = threadIdx.x;
    const long long row = blockIdx.x;
    const float* xr = X + row * C;
    u16t* yr = Yb + row * C;

    float m = -1e30f;
    for (int i = tid; i < C; i += 256) m = fmaxf(m, xr[i]);
    red[tid] = m; __syncthreads();
    for (int o = 128; o > 0; o >>= 1) {
        if (tid < o) red[tid] = fmaxf(red[tid], red[tid + o]);
        __syncthreads();
    }
    m = red[0]; __syncthreads();

    float s = 0.0f;
    for (int i = tid; i < C; i += 256) s += __expf(xr[i] - m);
    red[tid] = s; __syncthreads();
    for (int o = 128; o > 0; o >>= 1) {
        if (tid < o) red[tid] += red[tid + o];
        __syncthreads();
    }
    s = red[0]; __syncthreads();

    float inv = 1.0f / s;
    for (int i = tid; i < C; i += 256) yr[i] = f2bf(__expf(xr[i] - m) * inv);
}

// ---------------------------------------------------------------------------
// Fused residual add + LayerNorm. Optional f32 and bf16 outputs.
// ---------------------------------------------------------------------------
__global__ __launch_bounds__(256) void k_add_layernorm(
    const float* __restrict__ Aacc, const float* __restrict__ R,
    const float* __restrict__ g, const float* __restrict__ be,
    float* __restrict__ Yf, u16t* __restrict__ Yb, int C) {
    __shared__ float red[256];
    const int tid = threadIdx.x;
    const long long row = blockIdx.x;
    const float* ar = Aacc + row * C;
    const float* rr = R + row * C;

    float s = 0.0f, ss = 0.0f;
    for (int i = tid; i < C; i += 256) {
        float v = ar[i] + rr[i];
        s += v; ss += v * v;
    }
    red[tid] = s; __syncthreads();
    for (int o = 128; o > 0; o >>= 1) {
        if (tid < o) red[tid] += red[tid + o];
        __syncthreads();
    }
    s = red[0]; __syncthreads();
    red[tid] = ss; __syncthreads();
    for (int o = 128; o > 0; o >>= 1) {
        if (tid < o) red[tid] += red[tid + o];
        __syncthreads();
    }
    ss = red[0]; __syncthreads();

    float mu = s / (float)C;
    float var = ss / (float)C - mu * mu;
    float rstd = rsqrtf(var + 1e-5f);

    for (int i = tid; i < C; i += 256) {
        float v = (ar[i] + rr[i] - mu) * rstd * g[i] + be[i];
        if (Yf) Yf[row * C + i] = v;
        if (Yb) Yb[row * C + i] = f2bf(v);
    }
}

// ---------------------------------------------------------------------------
// Host launcher
// ---------------------------------------------------------------------------
extern "C" void kernel_launch(void* const* d_in, const int* in_sizes, int n_in,
                              void* d_out, int out_size, void* d_ws, size_t ws_size,
                              hipStream_t stream) {
    (void)in_sizes; (void)n_in; (void)out_size; (void)ws_size;

    const int Bb = 4, Nn = 1024, Dd = 1024, Hh = 16, DHh = 64, MLPd = 4096;
    const int BN = Bb * Nn;                    // 4096 rows
    const float sa_scale = 1.0f / 32.0f;       // 1/sqrt(1024)
    const float ca_alpha = 1.0f / 64.0f;       // (1/sqrt(64))^2

    const float* x    = (const float*)d_in[0];
    const float* enc  = (const float*)d_in[1];
    const float* sa_wq = (const float*)d_in[2];
    const float* sa_wk = (const float*)d_in[3];
    const float* sa_wv = (const float*)d_in[4];
    const float* sa_g = (const float*)d_in[5];
    const float* sa_b = (const float*)d_in[6];
    const float* ca_wq = (const float*)d_in[7];
    const float* ca_wk = (const float*)d_in[8];
    const float* ca_wv = (const float*)d_in[9];
    const float* ca_g = (const float*)d_in[10];
    const float* ca_b = (const float*)d_in[11];
    const float* w1   = (const float*)d_in[12];
    const float* b1   = (const float*)d_in[13];
    const float* w2   = (const float*)d_in[14];
    const float* b2   = (const float*)d_in[15];
    const float* ff_g = (const float*)d_in[16];
    const float* ff_b = (const float*)d_in[17];
    float* out = (float*)d_out;

    // --- bump allocator in d_ws (256B aligned) ---
    size_t cur = 0;
    auto alloc = [&](size_t bytes) -> void* {
        cur = (cur + 255) & ~(size_t)255;
        void* p = (char*)d_ws + cur;
        cur += bytes;
        return p;
    };
    const size_t DD = (size_t)Dd * Dd, BND = (size_t)BN * Dd, BNN = (size_t)BN * Nn;

    u16t* wqb  = (u16t*)alloc(DD * 2);
    u16t* wkb  = (u16t*)alloc(DD * 2);
    u16t* wvb  = (u16t*)alloc(DD * 2);
    u16t* cwqb = (u16t*)alloc(DD * 2);
    u16t* cwkb = (u16t*)alloc(DD * 2);
    u16t* cwvb = (u16t*)alloc(DD * 2);
    u16t* w1b  = (u16t*)alloc((size_t)Dd * MLPd * 2);
    u16t* w2b  = (u16t*)alloc((size_t)MLPd * Dd * 2);
    u16t* xb   = (u16t*)alloc(BND * 2);     // reused as sab after self-attn
    u16t* encb = (u16t*)alloc(BND * 2);     // reused as cab after cross-attn
    u16t* qb   = (u16t*)alloc(BND * 2);     // reused as qcb
    u16t* kb   = (u16t*)alloc(BND * 2);     // reused as kcb
    u16t* vb   = (u16t*)alloc(BND * 2);     // reused as vcb
    u16t* attnb = (u16t*)alloc(BNN * 2);
    u16t* tb   = (u16t*)alloc((size_t)BN * DHh * 2);
    u16t* h1b  = (u16t*)alloc((size_t)BN * MLPd * 2);

    float* dots   = (float*)alloc(BNN * 4);  // reused as h2
    float* sa_out = (float*)alloc(BND * 4);  // reused as caout
    float* ca_f   = (float*)alloc(BND * 4);

    u16t* sab = xb;
    u16t* cab = encb;
    u16t* qcb = qb; u16t* kcb = kb; u16t* vcb = vb;
    float* caout = sa_out;
    float* h2 = dots;

    auto cvt = [&](const float* s, u16t* d, long long n) {
        int blocks = (int)((n + 1023) / 1024);
        if (blocks > 4096) blocks = 4096;
        k_cvt_f32_bf16<<<blocks, 256, 0, stream>>>(s, d, n);
    };
    auto gemm = [&](bool transb, bool gelu,
                    const u16t* A, long long sA, int lda,
                    const u16t* Bp, long long sB, int ldb,
                    float* Cf, u16t* Cb, long long sC, int ldc,
                    int M, int Nc, int K, float alpha, const float* bias,
                    int batch) {
        dim3 g((Nc + 127) / 128, (M + 127) / 128, batch);
        if (transb)
            k_gemm_bf16<true, false><<<g, 256, 0, stream>>>(
                A, sA, lda, Bp, sB, ldb, Cf, Cb, sC, ldc, M, Nc, K, alpha, bias);
        else if (gelu)
            k_gemm_bf16<false, true><<<g, 256, 0, stream>>>(
                A, sA, lda, Bp, sB, ldb, Cf, Cb, sC, ldc, M, Nc, K, alpha, bias);
        else
            k_gemm_bf16<false, false><<<g, 256, 0, stream>>>(
                A, sA, lda, Bp, sB, ldb, Cf, Cb, sC, ldc, M, Nc, K, alpha, bias);
    };

    // --- convert inputs/weights to bf16 ---
    cvt(sa_wq, wqb, DD); cvt(sa_wk, wkb, DD); cvt(sa_wv, wvb, DD);
    cvt(ca_wq, cwqb, DD); cvt(ca_wk, cwkb, DD); cvt(ca_wv, cwvb, DD);
    cvt(w1, w1b, (long long)Dd * MLPd); cvt(w2, w2b, (long long)MLPd * Dd);
    cvt(x, xb, BND); cvt(enc, encb, BND);

    // === Self-attention (1 head, dim_head = D) ===
    gemm(false, false, xb, 0, Dd, wqb, 0, Dd, nullptr, qb, 0, Dd, BN, Dd, Dd, 1.0f, nullptr, 1);
    gemm(false, false, xb, 0, Dd, wkb, 0, Dd, nullptr, kb, 0, Dd, BN, Dd, Dd, 1.0f, nullptr, 1);
    gemm(false, false, xb, 0, Dd, wvb, 0, Dd, nullptr, vb, 0, Dd, BN, Dd, Dd, 1.0f, nullptr, 1);
    // dots[b] = scale * q[b] @ k[b]^T
    gemm(true, false, qb, (long long)Nn * Dd, Dd, kb, (long long)Nn * Dd, Dd,
         dots, nullptr, (long long)Nn * Nn, Nn, Nn, Nn, Dd, sa_scale, nullptr, Bb);
    k_softmax_rows<<<BN, 256, 0, stream>>>(dots, attnb, Nn);
    // sa_out[b] = attn[b] @ v[b]
    gemm(false, false, attnb, (long long)Nn * Nn, Nn, vb, (long long)Nn * Dd, Dd,
         sa_out, nullptr, (long long)Nn * Dd, Dd, Nn, Dd, Nn, 1.0f, nullptr, Bb);
    // sa = LN(sa_out + x)   (bf16 only; overwrites xb after last use)
    k_add_layernorm<<<BN, 256, 0, stream>>>(sa_out, x, sa_g, sa_b, nullptr, sab, Dd);

    // === Cross-attention (uses kkt symmetry: dots = s^2 * (qc@kc) @ kc^T) ===
    gemm(false, false, sab, 0, Dd, cwqb, 0, Dd, nullptr, qcb, 0, Dd, BN, Dd, Dd, 1.0f, nullptr, 1);
    gemm(false, false, encb, 0, Dd, cwkb, 0, Dd, nullptr, kcb, 0, Dd, BN, Dd, Dd, 1.0f, nullptr, 1);
    gemm(false, false, encb, 0, Dd, cwvb, 0, Dd, nullptr, vcb, 0, Dd, BN, Dd, Dd, 1.0f, nullptr, 1);

    for (int h = 0; h < Hh; ++h) {
        const u16t* kch = kcb + h * DHh;
        const u16t* vch = vcb + h * DHh;
        // t[b,h] = qc[b] @ kc[b,h]          [N x DH]
        gemm(false, false, qcb, (long long)Nn * Dd, Dd, kch, (long long)Nn * Dd, Dd,
             nullptr, tb, (long long)Nn * DHh, DHh, Nn, DHh, Dd, 1.0f, nullptr, Bb);
        // dots = s^2 * t @ kc[b,h]^T        [N x N]
        gemm(true, false, tb, (long long)Nn * DHh, DHh, kch, (long long)Nn * Dd, Dd,
             dots, nullptr, (long long)Nn * Nn, Nn, Nn, Nn, DHh, ca_alpha, nullptr, Bb);
        k_softmax_rows<<<BN, 256, 0, stream>>>(dots, attnb, Nn);
        // out slice = attn @ vc[b,h]   -> caout[:, h*DH : (h+1)*DH]
        gemm(false, false, attnb, (long long)Nn * Nn, Nn, vch, (long long)Nn * Dd, Dd,
             caout + h * DHh, nullptr, (long long)Nn * Dd, Dd, Nn, DHh, Nn, 1.0f, nullptr, Bb);
    }
    // ca = LN(out + enc_out)   (f32 for residual + bf16 for FFN; overwrites encb)
    k_add_layernorm<<<BN, 256, 0, stream>>>(caout, enc, ca_g, ca_b, ca_f, cab, Dd);

    // === FeedForward ===
    gemm(false, true, cab, 0, Dd, w1b, 0, MLPd, nullptr, h1b, 0, MLPd,
         BN, MLPd, Dd, 1.0f, b1, 1);
    gemm(false, false, h1b, 0, MLPd, w2b, 0, Dd, h2, nullptr, 0, Dd,
         BN, Dd, MLPd, 1.0f, b2, 1);
    k_add_layernorm<<<BN, 256, 0, stream>>>(h2, ca_f, ff_g, ff_b, out, nullptr, Dd);
}
